// Qwen3MoeMLP_61254823576069
// MI455X (gfx1250) — matrix operand order repaired and checked by reading, not compiled, not run on hardware
//
#include <hip/hip_runtime.h>

// MoE GLU MLP for MI455X (gfx1250, wave32, WMMA).
// Sparse top-2 execution; split-bf16 (hi/lo) WMMA for ~fp32 accuracy at bf16 rates.
// Round 4: fix async-to-LDS builtin argument types (as1/as3 v4i pointers, per the
// compiler diagnostic). Double-buffered LDS staging, one barrier per K-step,
// GLOBAL_LOAD_ASYNC_TO_LDS_B128 staging (ASYNCcnt-tracked, VGPR-free).
// Deterministic: ballot-scan compaction, no atomics, fixed combine order.

#define T_TOK 2048
#define HD    1024
#define ID    768
#define NE    8
#define ROWS_MAX (T_TOK * 2)   // top-2 => exactly 4096 (token,expert) pairs

typedef __attribute__((ext_vector_type(16))) __bf16 v16bf;
typedef __attribute__((ext_vector_type(8)))  __bf16 v8bf;
typedef __attribute__((ext_vector_type(8)))  float  v8f;
typedef __attribute__((ext_vector_type(4)))  int    v4i;

typedef __attribute__((address_space(1))) v4i gv4i;   // global-AS 16B vector
typedef __attribute__((address_space(3))) v4i lv4i;   // LDS-AS 16B vector

#if __has_builtin(__builtin_amdgcn_global_load_async_to_lds_b128)
#define USE_ASYNC_LDS 1
#else
#define USE_ASYNC_LDS 0
#endif

// ---- workspace layout (bytes). Requires ws_size >= ~114 MB. ----
#define WS_COUNTS 0u                        // 8  int
#define WS_BASES  64u                       // 9  int
#define WS_ROWMAP 1024u                     // T*E int            (65536 B)
#define WS_TOK    (WS_ROWMAP + 65536u)      // ROWS_MAX int       (16384 B)
#define WS_HHI    (WS_TOK + 16384u)         // ROWS_MAX*ID bf16   (6291456 B)
#define WS_HLO    (WS_HHI + 6291456u)
#define WS_Y      (WS_HLO + 6291456u)       // ROWS_MAX*HD f32    (16777216 B)
#define WS_XHI    (WS_Y + 16777216u)        // T*HD bf16          (4194304 B)
#define WS_XLO    (WS_XHI + 4194304u)
#define WS_GHI    (WS_XLO + 4194304u)       // E*ID*HD bf16       (12582912 B)
#define WS_GLO    (WS_GHI + 12582912u)
#define WS_UHI    (WS_GLO + 12582912u)
#define WS_ULO    (WS_UHI + 12582912u)
#define WS_DHI    (WS_ULO + 12582912u)      // E*HD*ID bf16       (12582912 B)
#define WS_DLO    (WS_DHI + 12582912u)
#define WS_ZERO   (WS_DLO + 12582912u)      // 4096 B zero page for padded rows
// end = WS_ZERO + 4096 = 113333248 B

static __device__ inline void split_bf16(float f, __bf16& hi, __bf16& lo) {
    // round-to-nearest-even truncation to bf16, residual to bf16
    unsigned u  = __builtin_bit_cast(unsigned, f);
    unsigned r  = u + 0x7fffu + ((u >> 16) & 1u);
    unsigned hb = r & 0xffff0000u;
    float    hf = __builtin_bit_cast(float, hb);
    hi = __builtin_bit_cast(__bf16, (unsigned short)(hb >> 16));
    float    lf = f - hf;
    unsigned u2 = __builtin_bit_cast(unsigned, lf);
    unsigned r2 = u2 + 0x7fffu + ((u2 >> 16) & 1u);
    lo = __builtin_bit_cast(__bf16, (unsigned short)(r2 >> 16));
}

static __device__ inline v16bf ld_frag2(const __bf16* p0, const __bf16* p1) {
    v8bf a = *(const v8bf*)p0;
    v8bf b = *(const v8bf*)p1;
    return __builtin_shufflevector(a, b, 0,1,2,3,4,5,6,7,8,9,10,11,12,13,14,15);
}

static __device__ inline v8f wmma_bf16(v16bf a, v16bf b, v8f c) {
    return __builtin_amdgcn_wmma_f32_16x16x32_bf16(false, a, false, b, (short)0, c, false, false);
}

#if USE_ASYNC_LDS
static __device__ inline void cp_async16(const __bf16* g, __bf16* l) {
    __builtin_amdgcn_global_load_async_to_lds_b128(
        (gv4i*)(g), (lv4i*)(l), 0, 0);
}
static __device__ inline void wait_async() {
#if __has_builtin(__builtin_amdgcn_s_wait_asynccnt)
    __builtin_amdgcn_s_wait_asynccnt(0);
#else
    asm volatile("s_wait_asynccnt 0x0" ::: "memory");
#endif
}
#endif

// ---------------- Kernel P: bulk f32 -> bf16 hi/lo split (bandwidth-bound pre-pass) ----------------
__global__ __launch_bounds__(256) void split_f32_kernel(
    const float* __restrict__ in, __bf16* __restrict__ hi, __bf16* __restrict__ lo, int n8)
{
    int g = blockIdx.x * 256 + threadIdx.x;
    if (g >= n8) return;
    const float4* p = (const float4*)(in + (size_t)g * 8);
    float4 a0 = p[0], a1 = p[1];
    float av[8] = {a0.x, a0.y, a0.z, a0.w, a1.x, a1.y, a1.z, a1.w};
    v8bf H, L;
    #pragma unroll
    for (int j = 0; j < 8; ++j) {
        __bf16 h_, l_; split_bf16(av[j], h_, l_);
        H[j] = h_; L[j] = l_;
    }
    *(v8bf*)(hi + (size_t)g * 8) = H;
    *(v8bf*)(lo + (size_t)g * 8) = L;
}

// ---------------- Kernel A: routing compaction (1 block, 8 waves) + zero page init ----------------
__global__ __launch_bounds__(256) void moe_route_compact(
    const float* __restrict__ routing, int* __restrict__ counts, int* __restrict__ bases,
    int* __restrict__ rowmap, int* __restrict__ tok_of_row, int* __restrict__ zero_page)
{
    const int tid = threadIdx.x, lane = tid & 31, e = tid >> 5;
    for (int i = tid; i < T_TOK * NE; i += 256) rowmap[i] = -1;
    for (int i = tid; i < 1024; i += 256) zero_page[i] = 0;
    __syncthreads();

    __shared__ int s_cnt[NE];
    __shared__ int s_base[NE + 1];

    int cnt = 0;
    for (int c = 0; c < T_TOK / 32; ++c) {
        int t = c * 32 + lane;
        float w = routing[t * NE + e];
        unsigned m = (unsigned)__ballot(w > 0.0f);
        cnt += __popc(m);
    }
    if (lane == 0) s_cnt[e] = cnt;
    __syncthreads();
    if (tid == 0) {
        int acc = 0;
        for (int i = 0; i < NE; ++i) { s_base[i] = acc; acc += s_cnt[i]; }
        s_base[NE] = acc;
        for (int i = 0; i < NE; ++i) { counts[i] = s_cnt[i]; bases[i] = s_base[i]; }
        bases[NE] = acc;
    }
    __syncthreads();

    const int base = s_base[e];
    int run = 0;
    for (int c = 0; c < T_TOK / 32; ++c) {
        int t = c * 32 + lane;
        float w = routing[t * NE + e];
        bool act = w > 0.0f;
        unsigned m = (unsigned)__ballot(act);
        int pos = run + __popc(m & ((1u << lane) - 1u));
        if (act) {
            int row = base + pos;
            tok_of_row[row] = t;
            rowmap[t * NE + e] = row;
        }
        run += __popc(m);
    }
}

// ---------------- Kernel B: gate/up GEMM + SiLU*up, output h as bf16 hi/lo ----------------
__global__ __launch_bounds__(256) void moe_gateup_kernel(
    const __bf16* __restrict__ x_hi, const __bf16* __restrict__ x_lo,
    const __bf16* __restrict__ g_hi, const __bf16* __restrict__ g_lo,
    const __bf16* __restrict__ u_hi, const __bf16* __restrict__ u_lo,
    const int* __restrict__ counts, const int* __restrict__ bases, const int* __restrict__ tok_of_row,
    const __bf16* __restrict__ zpage,
    __bf16* __restrict__ h_hi, __bf16* __restrict__ h_lo)
{
    __shared__ __align__(32) __bf16 sA_hi[2][64*32], sA_lo[2][64*32];
    __shared__ __align__(32) __bf16 sBg_hi[2][64*32], sBg_lo[2][64*32];
    __shared__ __align__(32) __bf16 sBu_hi[2][64*32], sBu_lo[2][64*32];

    const int e   = blockIdx.z;
    const int cnt = counts[e];
    const int m0  = blockIdx.x * 64;
    if (m0 >= cnt) return;                       // block-uniform: EXEC stays full for WMMA
    const int n0    = blockIdx.y * 64;           // intermediate-feature tile
    const int rbase = bases[e];

    const int tid = threadIdx.x, lane = tid & 31, wid = tid >> 5;
    const int wm = wid & 3, wn = wid >> 2;       // wave tile: rows 16*wm, cols 32*wn
    const int lr = tid >> 2, lk = (tid & 3) * 8; // staging: row lr (0..63), K chunk lk

    // Padded rows read the zero page (branch-free staging, async-friendly).
    const bool row_ok = (m0 + lr < cnt);
    const int  tok    = row_ok ? tok_of_row[rbase + m0 + lr] : 0;
    const size_t xoff = (size_t)tok * HD;
    const __bf16* xh_row = row_ok ? (x_hi + xoff + lk) : (zpage + lk);
    const __bf16* xl_row = row_ok ? (x_lo + xoff + lk) : (zpage + lk);
    const size_t woff = ((size_t)e * ID + (size_t)(n0 + lr)) * HD + lk;
    const __bf16* gh_row = g_hi + woff;
    const __bf16* gl_row = g_lo + woff;
    const __bf16* uh_row = u_hi + woff;
    const __bf16* ul_row = u_lo + woff;
    const int zstep = row_ok ? 32 : 0;           // zero page: don't advance with kk

    v8f accg[2] = { {0,0,0,0,0,0,0,0}, {0,0,0,0,0,0,0,0} };
    v8f accu[2] = { {0,0,0,0,0,0,0,0}, {0,0,0,0,0,0,0,0} };

    const int am = lane & 15, ah = lane >> 4;
    const int sidx = lr * 32 + lk;               // LDS staging slot

#if USE_ASYNC_LDS
    auto stage = [&](int b, int kk) {
        int ak = (kk / 32) * zstep;              // frozen at 0 for zero-page rows
        cp_async16(xh_row + ak,  &sA_hi[b][sidx]);
        cp_async16(xl_row + ak,  &sA_lo[b][sidx]);
        cp_async16(gh_row + kk,  &sBg_hi[b][sidx]);
        cp_async16(gl_row + kk,  &sBg_lo[b][sidx]);
        cp_async16(uh_row + kk,  &sBu_hi[b][sidx]);
        cp_async16(ul_row + kk,  &sBu_lo[b][sidx]);
    };
    stage(0, 0);
    wait_async();
    __syncthreads();
#else
    uint4 rg[6];
    auto ldreg = [&](int kk) {
        int ak = (kk / 32) * zstep;
        rg[0] = *(const uint4*)(xh_row + ak);
        rg[1] = *(const uint4*)(xl_row + ak);
        rg[2] = *(const uint4*)(gh_row + kk);
        rg[3] = *(const uint4*)(gl_row + kk);
        rg[4] = *(const uint4*)(uh_row + kk);
        rg[5] = *(const uint4*)(ul_row + kk);
    };
    auto streg = [&](int b) {
        *(uint4*)&sA_hi[b][sidx]  = rg[0];
        *(uint4*)&sA_lo[b][sidx]  = rg[1];
        *(uint4*)&sBg_hi[b][sidx] = rg[2];
        *(uint4*)&sBg_lo[b][sidx] = rg[3];
        *(uint4*)&sBu_hi[b][sidx] = rg[4];
        *(uint4*)&sBu_lo[b][sidx] = rg[5];
    };
    ldreg(0); streg(0);
    __syncthreads();
#endif

    for (int kk = 0; kk < HD; kk += 32) {
        const int cur = (kk >> 5) & 1, nxt = cur ^ 1;
        const bool more = (kk + 32 < HD);
#if USE_ASYNC_LDS
        if (more) stage(nxt, kk + 32);           // fire-and-forget into the other buffer
#else
        if (more) ldreg(kk + 32);                // issue global loads early
#endif

        // A fragments (16x32 bf16 layout: lane m=lane%16, K halves by lane>>4)
        const __bf16* pah = &sA_hi[cur][(16*wm + am)*32 + 16*ah];
        const __bf16* pal = &sA_lo[cur][(16*wm + am)*32 + 16*ah];
        v16bf a_hi = ld_frag2(pah, pah + 8);
        v16bf a_lo = ld_frag2(pal, pal + 8);

        #pragma unroll
        for (int s = 0; s < 2; ++s) {
            const int coff = (32*wn + 16*s + am)*32 + 16*ah;   // 32B-aligned contiguous run
            v16bf bg_h = *(const v16bf*)&sBg_hi[cur][coff];
            v16bf bg_l = *(const v16bf*)&sBg_lo[cur][coff];
            v16bf bu_h = *(const v16bf*)&sBu_hi[cur][coff];
            v16bf bu_l = *(const v16bf*)&sBu_lo[cur][coff];
            accg[s] = wmma_bf16(a_hi, bg_h, accg[s]);
            accg[s] = wmma_bf16(a_hi, bg_l, accg[s]);
            accg[s] = wmma_bf16(a_lo, bg_h, accg[s]);
            accu[s] = wmma_bf16(a_hi, bu_h, accu[s]);
            accu[s] = wmma_bf16(a_hi, bu_l, accu[s]);
            accu[s] = wmma_bf16(a_lo, bu_h, accu[s]);
        }

#if USE_ASYNC_LDS
        if (more) wait_async();                  // own wave's async copies landed in LDS
#else
        if (more) streg(nxt);                    // ds_store after compute (hides load latency)
#endif
        __syncthreads();
    }

    // GLU epilogue: h = silu(g)*u, store as bf16 hi/lo planes
    #pragma unroll
    for (int s = 0; s < 2; ++s) {
        #pragma unroll
        for (int r = 0; r < 8; ++r) {
            int ml = 16*wm + 8*ah + r;                 // C layout: lane>=16 -> M+8
            if (m0 + ml < cnt) {
                size_t row = (size_t)(rbase + m0 + ml);
                int col = n0 + 32*wn + 16*s + am;
                float g = accg[s][r], u = accu[s][r];
                float hv = (g / (1.0f + __expf(-g))) * u;
                __bf16 hh, hl; split_bf16(hv, hh, hl);
                h_hi[row * ID + col] = hh;
                h_lo[row * ID + col] = hl;
            }
        }
    }
}

// ---------------- Kernel C: down GEMM, y[row][H] = h_row @ down_w[e]^T ----------------
__global__ __launch_bounds__(256) void moe_down_kernel(
    const __bf16* __restrict__ h_hi, const __bf16* __restrict__ h_lo,
    const __bf16* __restrict__ d_hi, const __bf16* __restrict__ d_lo,
    const int* __restrict__ counts, const int* __restrict__ bases,
    const __bf16* __restrict__ zpage,
    float* __restrict__ y)
{
    __shared__ __align__(32) __bf16 sA_hi[2][64*32], sA_lo[2][64*32];
    __shared__ __align__(32) __bf16 sB_hi[2][64*32], sB_lo[2][64*32];

    const int e   = blockIdx.z;
    const int cnt = counts[e];
    const int m0  = blockIdx.x * 64;
    if (m0 >= cnt) return;
    const int n0    = blockIdx.y * 64;           // hidden-feature tile (H)
    const int rbase = bases[e];

    const int tid = threadIdx.x, lane = tid & 31, wid = tid >> 5;
    const int wm = wid & 3, wn = wid >> 2;
    const int lr = tid >> 2, lk = (tid & 3) * 8;
    const int am = lane & 15, ah = lane >> 4;
    const int sidx = lr * 32 + lk;

    const bool row_ok = (m0 + lr < cnt);
    const size_t arow = (size_t)(rbase + m0 + (row_ok ? lr : 0));
    const __bf16* ah_src = row_ok ? (h_hi + arow * ID + lk) : (zpage + lk);
    const __bf16* al_src = row_ok ? (h_lo + arow * ID + lk) : (zpage + lk);
    const size_t woff = ((size_t)e * HD + (size_t)(n0 + lr)) * ID + lk;
    const __bf16* bh_row = d_hi + woff;
    const __bf16* bl_row = d_lo + woff;
    const int zstep = row_ok ? 32 : 0;

    v8f acc[2] = { {0,0,0,0,0,0,0,0}, {0,0,0,0,0,0,0,0} };

#if USE_ASYNC_LDS
    auto stage = [&](int b, int kk) {
        int ak = (kk / 32) * zstep;
        cp_async16(ah_src + ak, &sA_hi[b][sidx]);
        cp_async16(al_src + ak, &sA_lo[b][sidx]);
        cp_async16(bh_row + kk, &sB_hi[b][sidx]);
        cp_async16(bl_row + kk, &sB_lo[b][sidx]);
    };
    stage(0, 0);
    wait_async();
    __syncthreads();
#else
    uint4 rg[4];
    auto ldreg = [&](int kk) {
        int ak = (kk / 32) * zstep;
        rg[0] = *(const uint4*)(ah_src + ak);
        rg[1] = *(const uint4*)(al_src + ak);
        rg[2] = *(const uint4*)(bh_row + kk);
        rg[3] = *(const uint4*)(bl_row + kk);
    };
    auto streg = [&](int b) {
        *(uint4*)&sA_hi[b][sidx] = rg[0];
        *(uint4*)&sA_lo[b][sidx] = rg[1];
        *(uint4*)&sB_hi[b][sidx] = rg[2];
        *(uint4*)&sB_lo[b][sidx] = rg[3];
    };
    ldreg(0); streg(0);
    __syncthreads();
#endif

    for (int kk = 0; kk < ID; kk += 32) {
        const int cur = (kk >> 5) & 1, nxt = cur ^ 1;
        const bool more = (kk + 32 < ID);
#if USE_ASYNC_LDS
        if (more) stage(nxt, kk + 32);
#else
        if (more) ldreg(kk + 32);
#endif

        const __bf16* pah = &sA_hi[cur][(16*wm + am)*32 + 16*ah];
        const __bf16* pal = &sA_lo[cur][(16*wm + am)*32 + 16*ah];
        v16bf a_hi = ld_frag2(pah, pah + 8);
        v16bf a_lo = ld_frag2(pal, pal + 8);

        #pragma unroll
        for (int s = 0; s < 2; ++s) {
            const int coff = (32*wn + 16*s + am)*32 + 16*ah;
            v16bf b_h = *(const v16bf*)&sB_hi[cur][coff];
            v16bf b_l = *(const v16bf*)&sB_lo[cur][coff];
            acc[s] = wmma_bf16(a_hi, b_h, acc[s]);
            acc[s] = wmma_bf16(a_hi, b_l, acc[s]);
            acc[s] = wmma_bf16(a_lo, b_h, acc[s]);
        }

#if USE_ASYNC_LDS
        if (more) wait_async();
#else
        if (more) streg(nxt);
#endif
        __syncthreads();
    }

    #pragma unroll
    for (int s = 0; s < 2; ++s) {
        #pragma unroll
        for (int r = 0; r < 8; ++r) {
            int ml = 16*wm + 8*ah + r;
            if (m0 + ml < cnt) {
                size_t row = (size_t)(rbase + m0 + ml);
                int col = n0 + 32*wn + 16*s + am;
                y[row * HD + col] = acc[s][r];
            }
        }
    }
}

// ---------------- Kernel D: deterministic combine over experts ----------------
__global__ __launch_bounds__(256) void moe_combine_kernel(
    const float* __restrict__ y, const int* __restrict__ rowmap,
    const float* __restrict__ routing, float* __restrict__ out)
{
    const int t = blockIdx.x;
    __shared__ int   rows[NE];
    __shared__ float wgt[NE];
    if (threadIdx.x < NE) {
        rows[threadIdx.x] = rowmap[t * NE + threadIdx.x];
        wgt[threadIdx.x]  = routing[t * NE + threadIdx.x];
    }
    __syncthreads();

    const int c = threadIdx.x;                   // 256 threads * float4 = 1024 cols
    float4 acc = {0.0f, 0.0f, 0.0f, 0.0f};
    #pragma unroll
    for (int e = 0; e < NE; ++e) {               // fixed ascending order => deterministic
        int r = rows[e];
        if (r >= 0) {
            float4 v = ((const float4*)(y + (size_t)r * HD))[c];
            float w = wgt[e];
            acc.x += w * v.x; acc.y += w * v.y; acc.z += w * v.z; acc.w += w * v.w;
        }
    }
    ((float4*)(out + (size_t)t * HD))[c] = acc;
}

extern "C" void kernel_launch(void* const* d_in, const int* in_sizes, int n_in,
                              void* d_out, int out_size, void* d_ws, size_t ws_size,
                              hipStream_t stream) {
    (void)in_sizes; (void)n_in; (void)out_size; (void)ws_size;
    const float* x       = (const float*)d_in[0];
    const float* gate_w  = (const float*)d_in[1];
    const float* up_w    = (const float*)d_in[2];
    const float* down_w  = (const float*)d_in[3];
    const float* routing = (const float*)d_in[4];
    // d_in[5] expert_select_count: recomputed on device (positions needed anyway)
    float* out = (float*)d_out;

    char* ws = (char*)d_ws;
    int*    counts = (int*)(ws + WS_COUNTS);
    int*    bases  = (int*)(ws + WS_BASES);
    int*    rowmap = (int*)(ws + WS_ROWMAP);
    int*    tokrow = (int*)(ws + WS_TOK);
    __bf16* h_hi   = (__bf16*)(ws + WS_HHI);
    __bf16* h_lo   = (__bf16*)(ws + WS_HLO);
    float*  y      = (float*)(ws + WS_Y);
    __bf16* x_hi   = (__bf16*)(ws + WS_XHI);
    __bf16* x_lo   = (__bf16*)(ws + WS_XLO);
    __bf16* g_hi   = (__bf16*)(ws + WS_GHI);
    __bf16* g_lo   = (__bf16*)(ws + WS_GLO);
    __bf16* u_hi   = (__bf16*)(ws + WS_UHI);
    __bf16* u_lo   = (__bf16*)(ws + WS_ULO);
    __bf16* d_hi   = (__bf16*)(ws + WS_DHI);
    __bf16* d_lo   = (__bf16*)(ws + WS_DLO);
    __bf16* zpage  = (__bf16*)(ws + WS_ZERO);

    const int n8_x = (T_TOK * HD) / 8;           // 262144
    const int n8_w = (NE * ID * HD) / 8;         // 786432 (same for gate/up/down)

    moe_route_compact<<<1, 256, 0, stream>>>(routing, counts, bases, rowmap, tokrow, (int*)zpage);
    split_f32_kernel<<<(n8_x + 255) / 256, 256, 0, stream>>>(x, x_hi, x_lo, n8_x);
    split_f32_kernel<<<(n8_w + 255) / 256, 256, 0, stream>>>(gate_w, g_hi, g_lo, n8_w);
    split_f32_kernel<<<(n8_w + 255) / 256, 256, 0, stream>>>(up_w, u_hi, u_lo, n8_w);
    split_f32_kernel<<<(n8_w + 255) / 256, 256, 0, stream>>>(down_w, d_hi, d_lo, n8_w);

    moe_gateup_kernel<<<dim3(T_TOK/64, ID/64, NE), 256, 0, stream>>>(
        x_hi, x_lo, g_hi, g_lo, u_hi, u_lo, counts, bases, tokrow, zpage, h_hi, h_lo);
    moe_down_kernel<<<dim3(T_TOK/64, HD/64, NE), 256, 0, stream>>>(
        h_hi, h_lo, d_hi, d_lo, counts, bases, zpage, y);
    moe_combine_kernel<<<T_TOK, 256, 0, stream>>>(y, rowmap, routing, out);
}
